// ARIMA_59373627900097
// MI455X (gfx1250) — matrix operand order repaired and checked by reading, not compiled, not run on hardware
//
#include <hip/hip_runtime.h>
#include <hip/hip_bf16.h>

typedef __attribute__((ext_vector_type(16))) _Float16 v16h;
typedef __attribute__((ext_vector_type(8)))  float    v8f;

#define NB 1024           // blocks in main kernel (also #partials in d_ws)
#define NT 256            // threads per block (8 waves on wave32)
#define WAVES_PER_BLOCK (NT / 32)

// Main kernel: each wave processes tiles of 16 consecutive timesteps.
// For tile base: window for output column n is yf[base+n+1 .. base+n+32].
// D1 = A1(16x32) x B(32x16): row0 = conv with wp, row1 = window sum.
// D2 = A2(16x32) x B^2     : row0 = window sum of squares.
__global__ __launch_bounds__(NT) void arima_wmma_main(
    const float* __restrict__ yf,    // [S]
    const float* __restrict__ arw,   // [32]
    const float* __restrict__ arb,   // [1]
    const float* __restrict__ revw,  // [1]
    const float* __restrict__ revb,  // [1]
    float* __restrict__ partial,     // [NB] block partial sums of err^2
    long T, long ntile)
{
    __shared__ float s_wp[32];
    __shared__ float s_red[NT];

    const int tid = threadIdx.x;

    // wp[j] = w[j] - w[j+1] (w[32]=0); telescoped differencing filter.
    if (tid < 32) {
        float wj  = arw[tid];
        float wj1 = (tid < 31) ? arw[tid + 1] : 0.0f;
        s_wp[tid] = wj - wj1;
    }
    __syncthreads();

    const float rw = revw[0];
    const float rb = revb[0];
    const float ab = arb[0];
    const float w0 = arw[0];               // sum of wp telescopes to w[0]
    const float inv_rw = 1.0f / (rw + 1e-10f);   // hoisted loop-invariant

    const int lane    = tid & 31;
    const int m       = lane & 15;   // A-matrix row / D-matrix column index
    const int hiHalf  = lane >> 4;   // 0: lanes 0-15, 1: lanes 16-31

    // Build A operands once (invariant across tiles).
    // 16-bit A 16x32 layout (contiguous, matching B): half h of lane l holds
    //   K = h + 16*(l>=16), row M = l%16.
    v16h A1, A2;
    #pragma unroll
    for (int h = 0; h < 16; ++h) {
        int K = h + 16 * hiHalf;
        float a1 = (m == 0) ? s_wp[K] : ((m == 1) ? 1.0f : 0.0f);
        A1[h] = (_Float16)a1;
        A2[h] = (m == 0) ? (_Float16)1.0f : (_Float16)0.0f;
    }

    // B 32x16 layout: lane l supplies column n = l%16; halves h=0..15 are
    // K = h for lanes 0-15 and K = 16+h for lanes 16-31.
    const int  n      = lane & 15;
    const long kb0    = hiHalf ? 16 : 0;
    const long gw     = (long)blockIdx.x * WAVES_PER_BLOCK + (tid >> 5);
    const long stride = (long)NB * WAVES_PER_BLOCK;

    float acc = 0.0f;

    for (long tile = gw; tile < ntile; tile += stride) {
        const long base = tile * 16;
        const float* src = yf + base + n + 1 + kb0;

        // Prefetch next tile for this wave (global_prefetch_b8).
        if (tile + stride < ntile)
            __builtin_prefetch(yf + (tile + stride) * 16 + n + 1 + kb0, 0, 1);

        v16h B, B2;
        #pragma unroll
        for (int h = 0; h < 16; ++h) {
            float x = src[h];
            B[h]  = (_Float16)x;
            B2[h] = (_Float16)(x * x);
        }

        v8f d1 = {};
        v8f d2 = {};
        d1 = __builtin_amdgcn_wmma_f32_16x16x32_f16(
                 false, A1, false, B,  (short)0, d1, false, false);
        d2 = __builtin_amdgcn_wmma_f32_16x16x32_f16(
                 false, A2, false, B2, (short)0, d2, false, false);

        // D 16x16 f32 layout: lanes 0-15 VGPR r = row r, col = lane.
        if (lane < 16) {
            long t = base + n;
            if (t < T) {
                float Cc   = d1[0];                 // conv with wp
                float S1v  = d1[1];                 // window sum
                float S2v  = d2[0];                 // window sum of squares
                float mean = S1v * (1.0f / 32.0f);
                float vpe  = S2v * (1.0f / 32.0f) - mean * mean + 1e-5f;
                // rs ~= 1/std, sd ~= std via one v_rsq_f32 (no sqrt/div chains)
                float rs   = __builtin_amdgcn_rsqf(vpe);
                float sd   = vpe * rs;
                float g    = rw * rs;
                // full-precision f32 loads for the sensitive scalar path
                float x30 = yf[t + 31];
                float x31 = yf[t + 32];
                float tgt = yf[t + 33];
                float ar     = g * Cc - (g * mean - rb) * w0 + ab;
                float pred_d = g * (x31 - x30) + ar;
                float pred   = (pred_d - rb) * inv_rw * sd + mean;
                float e = tgt - pred;
                acc += e * e;
            }
        }
    }

    // Deterministic block reduction -> one partial per block.
    s_red[tid] = acc;
    __syncthreads();
    for (int off = NT / 2; off > 0; off >>= 1) {
        if (tid < off) s_red[tid] += s_red[tid + off];
        __syncthreads();
    }
    if (tid == 0) partial[blockIdx.x] = s_red[0];
}

// Single-block finalize: deterministic tree over NB partials + head term.
__global__ __launch_bounds__(NT) void arima_finalize(
    const float* __restrict__ partial,
    const float* __restrict__ yf,
    float* __restrict__ out,
    long Sfull)
{
    __shared__ float s_red[NT];
    float acc = 0.0f;
    for (int i = threadIdx.x; i < NB; i += NT) acc += partial[i];
    s_red[threadIdx.x] = acc;
    __syncthreads();
    for (int off = NT / 2; off > 0; off >>= 1) {
        if (threadIdx.x < off) s_red[threadIdx.x] += s_red[threadIdx.x + off];
        __syncthreads();
    }
    if (threadIdx.x == 0) {
        float head = 0.0f;
        for (int i = 0; i < 33; ++i) { float v = yf[i]; head += v * v; }
        out[0] = (s_red[0] + head) / (float)Sfull;
    }
}

extern "C" void kernel_launch(void* const* d_in, const int* in_sizes, int n_in,
                              void* d_out, int out_size, void* d_ws, size_t ws_size,
                              hipStream_t stream) {
    const float* y    = (const float*)d_in[0];  // [1,S,1] fp32
    const float* arw  = (const float*)d_in[1];  // [1,32]
    const float* arb  = (const float*)d_in[2];  // [1]
    const float* revw = (const float*)d_in[3];  // [1]
    const float* revb = (const float*)d_in[4];  // [1]
    float* out = (float*)d_out;
    float* ws  = (float*)d_ws;                  // NB floats of scratch

    const long S  = (long)in_sizes[0];
    const long T  = S - 33;                     // t0 = max(P,Q)+1 = 33
    const long nt = (T + 15) / 16;              // 16 timesteps per WMMA tile

    arima_wmma_main<<<NB, NT, 0, stream>>>(y, arw, arb, revw, revb, ws, T, nt);
    arima_finalize<<<1, NT, 0, stream>>>(ws, y, out, S);
}
